// LocalTransformerBlock_63101659513221
// MI455X (gfx1250) — matrix operand order repaired and checked by reading, not compiled, not run on hardware
//
#include <hip/hip_runtime.h>
#include <hip/hip_bf16.h>
#include <math.h>

// ---------------------------------------------------------------------------
// LocalTransformerBlock for MI455X (gfx1250): wave32 + v_wmma_f32_16x16x32_f16
// All LDS staging is 128-bit; weights pre-transposed so no transpose scatter.
// Async direct-to-LDS loads (ASYNCcnt) used when the toolchain exposes them.
// ---------------------------------------------------------------------------

typedef __attribute__((ext_vector_type(16))) _Float16 v16h;
typedef __attribute__((ext_vector_type(8)))  float    v8f;

#define WMMA16(a, b, c) \
  __builtin_amdgcn_wmma_f32_16x16x32_f16(false, (a), false, (b), (short)0, (c), false, false)

// Problem constants (reference: B=2, N=8192, D=512, H=8, DH=64, WINDOW=512)
#define BATCH   2
#define SEQ     8192
#define DMODEL  512
#define HEADS_C 8
#define DH_C    64
#define BH_C    (BATCH * HEADS_C)
#define MROWS   (BATCH * SEQ)          // 16384
#define WS_C    512
#define HID_C   1365                   // int(512*4*2/3)
#define HID2_C  2730                   // 2*HID
#define HIDP_C  1376                   // HID padded to multiple of 32

// -------- 16B global->LDS copy: async (ASYNCcnt) if available ---------------
#if defined(__has_builtin)
#if __has_builtin(__builtin_amdgcn_global_load_async_to_lds_b128)
#define HAVE_ASYNC_LDS 1
#endif
#endif

#ifdef HAVE_ASYNC_LDS
typedef int v4i_vs __attribute__((vector_size(16)));
typedef __attribute__((address_space(1))) v4i_vs gv4i;  // global int4
typedef __attribute__((address_space(3))) v4i_vs lv4i;  // LDS int4
__device__ __forceinline__ void lds_copy16(void* lds, const void* g) {
  // flat->AS1: same 64-bit address; flat->AS3: LDS offset is addr[31:0]
  __builtin_amdgcn_global_load_async_to_lds_b128(
      (gv4i*)(uintptr_t)g,
      (lv4i*)(uint32_t)(uintptr_t)lds, 0, 0);
}
__device__ __forceinline__ void lds_copy_wait() {
  asm volatile("s_wait_asynccnt 0x0" ::: "memory");
}
#else
__device__ __forceinline__ void lds_copy16(void* lds, const void* g) {
  *(uint4*)lds = *(const uint4*)g;
}
__device__ __forceinline__ void lds_copy_wait() {}
#endif

// -------- fragment load helpers (16B-aligned chunks) ------------------------
__device__ __forceinline__ v16h ld_frag_split(const _Float16* p0, const _Float16* p1) {
  v16h r;
  ((uint4*)&r)[0] = *(const uint4*)p0;
  ((uint4*)&r)[1] = *(const uint4*)p1;
  return r;
}
__device__ __forceinline__ v16h ld_frag_contig(const _Float16* p) {
  v16h r;
  ((uint4*)&r)[0] = *(const uint4*)p;
  ((uint4*)&r)[1] = *(const uint4*)(p + 8);
  return r;
}

// -------- wave32 shuffle reductions -----------------------------------------
__device__ __forceinline__ float halfwave_max(float v) {
  #pragma unroll
  for (int d = 1; d < 16; d <<= 1) v = fmaxf(v, __shfl_xor(v, d, 32));
  return v;
}
__device__ __forceinline__ float halfwave_sum(float v) {
  #pragma unroll
  for (int d = 1; d < 16; d <<= 1) v += __shfl_xor(v, d, 32);
  return v;
}
__device__ __forceinline__ float wave_sum(float v) {
  #pragma unroll
  for (int d = 1; d < 32; d <<= 1) v += __shfl_xor(v, d, 32);
  return v;
}

// ---------------------------------------------------------------------------
// Weight prep: fp32 -> f16, transposed to [N][K] so GEMM B-staging is b128.
// ---------------------------------------------------------------------------
// src [R][C] fp32 -> dst [C][R] f16 (coalesced writes)
__global__ __launch_bounds__(256) void transpose_f16_kernel(const float* __restrict__ src,
                                                            _Float16* __restrict__ dst,
                                                            int R, int C) {
  int i = blockIdx.x * 256 + threadIdx.x;
  if (i >= R * C) return;
  int c = i / R, r = i - c * R;
  dst[i] = (_Float16)src[(size_t)r * C + c];
}

// W1 [512][2730] -> aT [1376][512], gT [1376][512] (pad rows zero)
__global__ __launch_bounds__(256) void w1_split_T_kernel(const float* __restrict__ w1,
                                                         _Float16* __restrict__ aT,
                                                         _Float16* __restrict__ gT) {
  int i = blockIdx.x * 256 + threadIdx.x;
  if (i >= HIDP_C * DMODEL) return;
  int col = i / DMODEL, k = i - col * DMODEL;
  aT[i] = (col < HID_C) ? (_Float16)w1[(size_t)k * HID2_C + col] : (_Float16)0.f;
  gT[i] = (col < HID_C) ? (_Float16)w1[(size_t)k * HID2_C + col + HID_C] : (_Float16)0.f;
}

// W2 [1365][512] -> w2T [512][1376] (pad cols zero)
__global__ __launch_bounds__(256) void w2_T_pad_kernel(const float* __restrict__ w2,
                                                       _Float16* __restrict__ dst) {
  int i = blockIdx.x * 256 + threadIdx.x;
  if (i >= DMODEL * HIDP_C) return;
  int n = i / HIDP_C, k = i - n * HIDP_C;
  dst[i] = (k < HID_C) ? (_Float16)w2[(size_t)k * DMODEL + n] : (_Float16)0.f;
}

// ---------------------------------------------------------------------------
// LayerNorm (row of 512) -> f16
// ---------------------------------------------------------------------------
__global__ __launch_bounds__(128) void layernorm_f16_kernel(const float* __restrict__ x,
                                                            const float* __restrict__ g,
                                                            const float* __restrict__ b,
                                                            _Float16* __restrict__ out) {
  const int row = blockIdx.x;
  const int tid = threadIdx.x, wv = tid >> 5, lane = tid & 31;
  const float* xr = x + (size_t)row * DMODEL;
  float vals[4], s = 0.f, ss = 0.f;
  #pragma unroll
  for (int i = 0; i < 4; ++i) {
    float v = xr[tid + i * 128];
    vals[i] = v; s += v; ss += v * v;
  }
  s = wave_sum(s); ss = wave_sum(ss);
  __shared__ float ps[2][4];
  if (lane == 0) { ps[0][wv] = s; ps[1][wv] = ss; }
  __syncthreads();
  s = ps[0][0] + ps[0][1] + ps[0][2] + ps[0][3];
  ss = ps[1][0] + ps[1][1] + ps[1][2] + ps[1][3];
  const float mean = s * (1.f / DMODEL);
  const float var  = ss * (1.f / DMODEL) - mean * mean;
  const float inv  = rsqrtf(var + 1e-5f);
  _Float16* orow = out + (size_t)row * DMODEL;
  #pragma unroll
  for (int i = 0; i < 4; ++i) {
    int c = tid + i * 128;
    orow[c] = (_Float16)((vals[i] - mean) * inv * g[c] + b[c]);
  }
}

// ---------------------------------------------------------------------------
// Generic WMMA GEMM: C[M,N] = A[M,K]f16 @ Bt[N,K]f16 (B pre-transposed)
//   EPI 0: C -> fp32 out | EPI 1: +resid | EPI 2: +bias +resid
// Block: 128 threads (4 waves), tile 64x64, K-step 32, pure b128 staging.
// ---------------------------------------------------------------------------
template <int EPI>
__global__ __launch_bounds__(128) void gemm_wmma_kernel(const _Float16* __restrict__ A,
                                                        const _Float16* __restrict__ Bt,
                                                        const float* __restrict__ resid,
                                                        const float* __restrict__ bias,
                                                        float* __restrict__ Cout,
                                                        int M, int N, int K) {
  __shared__ _Float16 As[64][32];   // [m][k]
  __shared__ _Float16 Bts[64][32];  // [n][k]
  const int m0 = blockIdx.x * 64, n0 = blockIdx.y * 64;
  const int tid = threadIdx.x, wv = tid >> 5, lane = tid & 31;
  const int lo = lane & 15, hi = lane >> 4;

  v8f acc[4] = {v8f{}, v8f{}, v8f{}, v8f{}};

  for (int k0 = 0; k0 < K; k0 += 32) {
    const int r = tid >> 1, c = (tid & 1) * 16;
    lds_copy16(&As[r][c],  A  + (size_t)(m0 + r) * K + k0 + c);
    lds_copy16(&Bts[r][c], Bt + (size_t)(n0 + r) * K + k0 + c);
    lds_copy_wait();
    __syncthreads();

    v16h af = ld_frag_contig(&As[wv * 16 + lo][hi * 16]);
    #pragma unroll
    for (int t = 0; t < 4; ++t) {
      v16h bf = ld_frag_contig(&Bts[t * 16 + lo][hi * 16]);
      acc[t] = WMMA16(af, bf, acc[t]);
    }
    __syncthreads();
  }

  #pragma unroll
  for (int t = 0; t < 4; ++t) {
    int col = n0 + t * 16 + lo;
    float bval = (EPI == 2) ? bias[col] : 0.f;
    #pragma unroll
    for (int r = 0; r < 8; ++r) {
      int row = m0 + wv * 16 + hi * 8 + r;
      float v = acc[t][r] + bval;
      if (EPI >= 1) v += resid[(size_t)row * N + col];
      Cout[(size_t)row * N + col] = v;
    }
  }
}

// ---------------------------------------------------------------------------
// QKV post: l2norm * scale, rotary-xpos -> q16/k16 [bh][n][d], vT [bh][d][n]
// One wave per (bh, n): lane owns dims d=lane and d=lane+32.
// ---------------------------------------------------------------------------
__global__ __launch_bounds__(128) void qkv_post_kernel(const float* __restrict__ qkv,
                                                       const float* __restrict__ q_scale,
                                                       const float* __restrict__ k_scale,
                                                       _Float16* __restrict__ q16,
                                                       _Float16* __restrict__ k16,
                                                       _Float16* __restrict__ vT) {
  const int wv = threadIdx.x >> 5, lane = threadIdx.x & 31;
  const int n = blockIdx.x * 4 + wv;
  const int bh = blockIdx.y;
  const int b = bh >> 3, h = bh & 7;
  const size_t base = ((size_t)b * SEQ + n) * (3 * DMODEL) + h * DH_C;

  float q0 = qkv[base + lane],              q1 = qkv[base + lane + 32];
  float k0 = qkv[base + DMODEL + lane],     k1 = qkv[base + DMODEL + lane + 32];
  float v0 = qkv[base + 2 * DMODEL + lane], v1 = qkv[base + 2 * DMODEL + lane + 32];

  float qs = wave_sum(q0 * q0 + q1 * q1);
  float ks = wave_sum(k0 * k0 + k1 * k1);
  float qinv = 1.f / fmaxf(sqrtf(qs), 1e-12f);
  float kinv = 1.f / fmaxf(sqrtf(ks), 1e-12f);
  q0 *= qinv * q_scale[lane]; q1 *= qinv * q_scale[lane + 32];
  k0 *= kinv * k_scale[lane]; k1 *= kinv * k_scale[lane + 32];

  const float t = (float)n;
  const float invf = powf(10000.f, -2.f * (float)lane / (float)DH_C);
  const float fr = t * invf;
  const float c = cosf(fr), s = sinf(fr);
  const float sv = (2.f * (float)lane + 0.4f * DH_C) / (1.4f * DH_C);
  const float power = (t - (float)(SEQ / 2)) / (float)(WS_C / 2);
  const float xs = powf(sv, power);
  const float xi = 1.f / xs;

  float qr0 = (q0 * c - q1 * s) * xs;
  float qr1 = (q1 * c + q0 * s) * xs;
  float kr0 = (k0 * c - k1 * s) * xi;
  float kr1 = (k1 * c + k0 * s) * xi;

  const size_t ob = ((size_t)bh * SEQ + n) * DH_C;
  q16[ob + lane] = (_Float16)qr0; q16[ob + lane + 32] = (_Float16)qr1;
  k16[ob + lane] = (_Float16)kr0; k16[ob + lane + 32] = (_Float16)kr1;
  // V transposed: [bh][d][n] so attention stages it with b128 copies
  vT[((size_t)bh * DH_C + lane)      * SEQ + n] = (_Float16)v0;
  vT[((size_t)bh * DH_C + lane + 32) * SEQ + n] = (_Float16)v1;
}

// ---------------------------------------------------------------------------
// Local windowed flash attention.
// Grid: (qtile=8, window=16, bh=16). Block: 128 threads (4 waves).
// Band pruning: only chunks [max(cstart,qt), qt+8] intersect j in [i-512, i].
// Interior chunks take a uniform no-mask fast path.
// ---------------------------------------------------------------------------
__global__ __launch_bounds__(128) void local_attn_kernel(const _Float16* __restrict__ q16,
                                                         const _Float16* __restrict__ k16,
                                                         const _Float16* __restrict__ vT,
                                                         _Float16* __restrict__ out16) {
  __shared__ _Float16 Ks[64][64];       // [key][d]   (B-layout for S = q @ k^T)
  __shared__ _Float16 Vts[64][64];      // [d][key]   (B-layout for P @ V)
  __shared__ _Float16 Ps[4][16][64];    // per-wave probabilities (A-layout source)

  const int tid = threadIdx.x, wv = tid >> 5, lane = tid & 31;
  const int lo = lane & 15, hi = lane >> 4;
  const int qt = blockIdx.x, win = blockIdx.y, bh = blockIdx.z;
  const int qrow0 = win * WS_C + qt * 64 + wv * 16;

  const _Float16* qb = q16 + ((size_t)bh * SEQ + qrow0) * DH_C;
  v16h qa[2];
  #pragma unroll
  for (int s = 0; s < 2; ++s)
    qa[s] = ld_frag_contig(qb + lo * DH_C + s * 32 + hi * 16);

  v8f oacc[4] = {v8f{}, v8f{}, v8f{}, v8f{}};
  float m_run[8], l_run[8];
  #pragma unroll
  for (int r = 0; r < 8; ++r) { m_run[r] = -3.0e38f; l_run[r] = 0.f; }

  // chunk range intersecting the valid band for this 64-row q-tile
  const int cbeg0 = (win == 0) ? 8 : 0;
  const int cbeg = (qt > cbeg0) ? qt : cbeg0;
  const int cend = (qt + 8 < 15) ? (qt + 8) : 15;

  for (int chunk = cbeg; chunk <= cend; ++chunk) {
    const int kn0 = (win - 1) * WS_C + chunk * 64;   // global key base (>= 0 here)
    __syncthreads();
    // stage K (row-major) and V^T: pure 16B copies, 8/thread
    #pragma unroll
    for (int i2 = 0; i2 < 4; ++i2) {
      int slot = tid + i2 * 128;
      int row = slot >> 3, ch = (slot & 7) * 8;
      lds_copy16(&Ks[row][ch],  k16 + ((size_t)bh * SEQ + kn0 + row) * DH_C + ch);
      lds_copy16(&Vts[row][ch], vT  + ((size_t)bh * DH_C + row) * SEQ + kn0 + ch);
    }
    lds_copy_wait();
    __syncthreads();

    // S = q @ k^T  (4 col-tiles x 2 K-steps = 8 wmma)
    v8f sacc[4] = {v8f{}, v8f{}, v8f{}, v8f{}};
    #pragma unroll
    for (int t = 0; t < 4; ++t)
      #pragma unroll
      for (int s = 0; s < 2; ++s) {
        v16h bf = ld_frag_contig(&Ks[t * 16 + lo][s * 32 + hi * 16]);
        sacc[t] = WMMA16(qa[s], bf, sacc[t]);
      }

    // mask + scale (QK_SCALE = 8); uniform fast path for interior chunks
    const bool full_valid = (kn0 + 63 <= qrow0) && (kn0 >= qrow0 + 15 - WS_C);
    if (full_valid) {
      #pragma unroll
      for (int t = 0; t < 4; ++t)
        #pragma unroll
        for (int r = 0; r < 8; ++r) sacc[t][r] *= 8.0f;
    } else {
      #pragma unroll
      for (int t = 0; t < 4; ++t) {
        const int j = kn0 + t * 16 + lo;
        #pragma unroll
        for (int r = 0; r < 8; ++r) {
          const int i = qrow0 + hi * 8 + r;
          const bool valid = (j <= i) && (j >= i - WS_C);
          sacc[t][r] = valid ? sacc[t][r] * 8.0f : -3.0e38f;
        }
      }
    }

    // online softmax update
    #pragma unroll
    for (int r = 0; r < 8; ++r) {
      float mx = sacc[0][r];
      #pragma unroll
      for (int t = 1; t < 4; ++t) mx = fmaxf(mx, sacc[t][r]);
      mx = halfwave_max(mx);
      const float mnew = fmaxf(m_run[r], mx);
      const float fac = expf(m_run[r] - mnew);
      float rs = 0.f;
      #pragma unroll
      for (int t = 0; t < 4; ++t) {
        float p = expf(sacc[t][r] - mnew);
        sacc[t][r] = p; rs += p;
      }
      rs = halfwave_sum(rs);
      l_run[r] = l_run[r] * fac + rs;
      m_run[r] = mnew;
      #pragma unroll
      for (int t = 0; t < 4; ++t) oacc[t][r] *= fac;
    }

    // P (f16) through LDS: C layout -> A layout
    #pragma unroll
    for (int t = 0; t < 4; ++t)
      #pragma unroll
      for (int r = 0; r < 8; ++r)
        Ps[wv][hi * 8 + r][t * 16 + lo] = (_Float16)sacc[t][r];
    __syncthreads();

    // O += P @ V  (2 K-steps x 4 d-tiles = 8 wmma)
    #pragma unroll
    for (int s = 0; s < 2; ++s) {
      v16h pf = ld_frag_contig(&Ps[wv][lo][s * 32 + hi * 16]);
      #pragma unroll
      for (int t = 0; t < 4; ++t) {
        v16h vf = ld_frag_contig(&Vts[t * 16 + lo][s * 32 + hi * 16]);
        oacc[t] = WMMA16(pf, vf, oacc[t]);
      }
    }
  }

  // normalize and write out as [B][n][h*64+d] f16
  const int b = bh >> 3, h = bh & 7;
  #pragma unroll
  for (int r = 0; r < 8; ++r) {
    const float linv = 1.f / fmaxf(l_run[r], 1e-20f);
    const int row = qrow0 + hi * 8 + r;
    #pragma unroll
    for (int t = 0; t < 4; ++t)
      out16[((size_t)b * SEQ + row) * DMODEL + h * DH_C + t * 16 + lo] =
          (_Float16)(oacc[t][r] * linv);
  }
}

// ---------------------------------------------------------------------------
// FF1 gated GEMM with pre-transposed padded halves aT/gT [1376][512]:
// act[c] = (h@W1a[c]+b1[c]) * gelu(h@W1g[c]+b1[c+HID]), f16, stride HIDP.
// ---------------------------------------------------------------------------
__global__ __launch_bounds__(128) void gemm_ff1_gated_kernel(const _Float16* __restrict__ A,
                                                             const _Float16* __restrict__ aT,
                                                             const _Float16* __restrict__ gT,
                                                             const float* __restrict__ b1,
                                                             _Float16* __restrict__ act) {
  __shared__ _Float16 As[64][32];
  __shared__ _Float16 Bta[64][32];
  __shared__ _Float16 Btg[64][32];
  const int m0 = blockIdx.x * 64, n0 = blockIdx.y * 64;
  const int tid = threadIdx.x, wv = tid >> 5, lane = tid & 31;
  const int lo = lane & 15, hi = lane >> 4;
  const int K = DMODEL;

  v8f acca[4] = {v8f{}, v8f{}, v8f{}, v8f{}};
  v8f accg[4] = {v8f{}, v8f{}, v8f{}, v8f{}};

  for (int k0 = 0; k0 < K; k0 += 32) {
    const int r = tid >> 1, c = (tid & 1) * 16;
    lds_copy16(&As[r][c],  A  + (size_t)(m0 + r) * K + k0 + c);
    lds_copy16(&Bta[r][c], aT + (size_t)(n0 + r) * K + k0 + c);
    lds_copy16(&Btg[r][c], gT + (size_t)(n0 + r) * K + k0 + c);
    lds_copy_wait();
    __syncthreads();

    v16h af = ld_frag_contig(&As[wv * 16 + lo][hi * 16]);
    #pragma unroll
    for (int t = 0; t < 4; ++t) {
      v16h bfa = ld_frag_contig(&Bta[t * 16 + lo][hi * 16]);
      v16h bfg = ld_frag_contig(&Btg[t * 16 + lo][hi * 16]);
      acca[t] = WMMA16(af, bfa, acca[t]);
      accg[t] = WMMA16(af, bfg, accg[t]);
    }
    __syncthreads();
  }

  #pragma unroll
  for (int t = 0; t < 4; ++t) {
    const int col = n0 + t * 16 + lo;
    const bool in = (col < HID_C);
    const float ba = in ? b1[col] : 0.f;
    const float bg = in ? b1[col + HID_C] : 0.f;
    #pragma unroll
    for (int r = 0; r < 8; ++r) {
      const int row = m0 + wv * 16 + hi * 8 + r;
      if (col < HIDP_C) {
        float va = 0.f;
        if (in) {
          const float a = acca[t][r] + ba;
          const float g = accg[t][r] + bg;
          va = a * 0.5f * g * (1.0f + erff(g * 0.70710678118654752f));
        }
        act[(size_t)row * HIDP_C + col] = (_Float16)va;   // pads written as 0
      }
    }
  }
}

// ---------------------------------------------------------------------------
// Host launcher
// ---------------------------------------------------------------------------
extern "C" void kernel_launch(void* const* d_in, const int* in_sizes, int n_in,
                              void* d_out, int out_size, void* d_ws, size_t ws_size,
                              hipStream_t stream) {
  (void)in_sizes; (void)n_in; (void)out_size; (void)ws_size;
  const float* x        = (const float*)d_in[0];
  const float* ln1_g    = (const float*)d_in[1];
  const float* ln1_b    = (const float*)d_in[2];
  const float* Wqkv     = (const float*)d_in[3];
  const float* q_scale  = (const float*)d_in[4];
  const float* k_scale  = (const float*)d_in[5];
  const float* Wout     = (const float*)d_in[6];
  const float* ln2_g    = (const float*)d_in[7];
  const float* ln2_b    = (const float*)d_in[8];
  const float* ff_W1    = (const float*)d_in[9];
  const float* ff_b1    = (const float*)d_in[10];
  const float* ff_W2    = (const float*)d_in[11];
  const float* ff_b2    = (const float*)d_in[12];
  float* out = (float*)d_out;

  // ---- workspace carve-up (256B aligned) ----
  char* base = (char*)d_ws;
  size_t off = 0;
  auto carve = [&](size_t bytes) -> char* {
    char* p = base + off;
    off = (off + bytes + 255) & ~(size_t)255;
    return p;
  };
  _Float16* wqkvT = (_Float16*)carve((size_t)3 * DMODEL * DMODEL * 2);   // [1536][512]
  _Float16* woutT = (_Float16*)carve((size_t)DMODEL * DMODEL * 2);       // [512][512]
  _Float16* w1aT  = (_Float16*)carve((size_t)HIDP_C * DMODEL * 2);       // [1376][512]
  _Float16* w1gT  = (_Float16*)carve((size_t)HIDP_C * DMODEL * 2);       // [1376][512]
  _Float16* w2T   = (_Float16*)carve((size_t)DMODEL * HIDP_C * 2);       // [512][1376]
  _Float16* h16   = (_Float16*)carve((size_t)MROWS * DMODEL * 2);        // reused for LN2
  float*    qkvf  = (float*)   carve((size_t)MROWS * 3 * DMODEL * 4);
  _Float16* q16   = (_Float16*)carve((size_t)BH_C * SEQ * DH_C * 2);
  _Float16* k16   = (_Float16*)carve((size_t)BH_C * SEQ * DH_C * 2);
  _Float16* vT    = (_Float16*)carve((size_t)BH_C * DH_C * SEQ * 2);
  _Float16* ao16  = (_Float16*)carve((size_t)MROWS * DMODEL * 2);
  float*    x1    = (float*)   carve((size_t)MROWS * DMODEL * 4);
  _Float16* act16 = (_Float16*)carve((size_t)MROWS * HIDP_C * 2);

  // 1) weight prep (convert + transpose [+ pad])
  {
    int n;
    n = DMODEL * 3 * DMODEL;
    transpose_f16_kernel<<<(n + 255) / 256, 256, 0, stream>>>(Wqkv, wqkvT, DMODEL, 3 * DMODEL);
    n = DMODEL * DMODEL;
    transpose_f16_kernel<<<(n + 255) / 256, 256, 0, stream>>>(Wout, woutT, DMODEL, DMODEL);
    n = HIDP_C * DMODEL;
    w1_split_T_kernel<<<(n + 255) / 256, 256, 0, stream>>>(ff_W1, w1aT, w1gT);
    n = DMODEL * HIDP_C;
    w2_T_pad_kernel<<<(n + 255) / 256, 256, 0, stream>>>(ff_W2, w2T);
  }

  // 2) LN1 -> h16
  layernorm_f16_kernel<<<MROWS, 128, 0, stream>>>(x, ln1_g, ln1_b, h16);

  // 3) QKV GEMM: (16384 x 512) @ (512 x 1536) -> fp32
  gemm_wmma_kernel<0><<<dim3(MROWS / 64, (3 * DMODEL) / 64), 128, 0, stream>>>(
      h16, wqkvT, nullptr, nullptr, qkvf, MROWS, 3 * DMODEL, DMODEL);

  // 4) l2norm + scales + rotary-xpos -> q16/k16/vT
  qkv_post_kernel<<<dim3(SEQ / 4, BH_C), 128, 0, stream>>>(qkvf, q_scale, k_scale,
                                                           q16, k16, vT);

  // 5) local windowed attention -> ao16 [B][n][H*DH]
  local_attn_kernel<<<dim3(WS_C / 64, SEQ / WS_C, BH_C), 128, 0, stream>>>(
      q16, k16, vT, ao16);

  // 6) output projection + residual: x1 = x + ao @ Wout
  gemm_wmma_kernel<1><<<dim3(MROWS / 64, DMODEL / 64), 128, 0, stream>>>(
      ao16, woutT, x, nullptr, x1, MROWS, DMODEL, DMODEL);

  // 7) LN2 -> h16 (reuse)
  layernorm_f16_kernel<<<MROWS, 128, 0, stream>>>(x1, ln2_g, ln2_b, h16);

  // 8) FF1 gated GEMM -> act16 (padded to 1376 cols)
  gemm_ff1_gated_kernel<<<dim3(MROWS / 64, HIDP_C / 64), 128, 0, stream>>>(
      h16, w1aT, w1gT, ff_b1, act16);

  // 9) FF2 GEMM + bias + residual -> out
  gemm_wmma_kernel<2><<<dim3(MROWS / 64, DMODEL / 64), 128, 0, stream>>>(
      act16, w2T, x1, ff_b2, out, MROWS, DMODEL, HIDP_C);
}
